// KernelizedHeadAttention_31430570672494
// MI455X (gfx1250) — matrix operand order repaired and checked by reading, not compiled, not run on hardware
//
#include <hip/hip_runtime.h>
#include <math.h>

typedef __attribute__((ext_vector_type(16))) __bf16 v16bf;
typedef __attribute__((ext_vector_type(8)))  float  v8f;

#define SEQ   2048
#define DMOD  2048
#define NH    16
#define DHEAD 128
#define DHID_ 128
#define DKER_ 64
#define EPSV  1e-6f

static __device__ __forceinline__ v8f wmma_bf16(v16bf a, v16bf b, v8f c) {
  // D = A(16x32 bf16) * B(32x16 bf16) + C(16x16 f32)
  return __builtin_amdgcn_wmma_f32_16x16x32_bf16(false, a, false, b, (short)0, c,
                                                 false, false);
}

static __device__ __forceinline__ v8f zero8() {
  v8f z;
#pragma unroll
  for (int i = 0; i < 8; ++i) z[i] = 0.0f;
  return z;
}

static __device__ __forceinline__ float gelu_exact(float x) {
  return 0.5f * x * (1.0f + erff(x * 0.7071067811865475f));
}

// ---------------------------------------------------------------------------
// Repack nmat row-major [Kdim][Ndim] fp32 matrices into WMMA B-fragment bf16.
// B-frag (ISA 7.12.2): lane l (0..15) = column, half hi selects K octets hi and hi+2:
//   elem j -> K = kt*32 + hi*8 + (j<8 ? j : j+8), N = nt*16 + l
// dst flat: (((h*KT + kt)*NT + nt)*32 + lane)*16 + j
// ---------------------------------------------------------------------------
__global__ void repack_b_kernel(const float* __restrict__ src,
                                __bf16* __restrict__ dst, int Kdim, int Ndim,
                                int nmat) {
  const int total = nmat * Kdim * Ndim;
  const int idx = blockIdx.x * blockDim.x + threadIdx.x;
  if (idx >= total) return;
  const int j = idx & 15;
  const int lane = (idx >> 4) & 31;
  const int f = idx >> 9;
  const int NT = Ndim >> 4;
  const int KT = Kdim >> 5;
  const int nt = f % NT;
  const int kt = (f / NT) % KT;
  const int h = f / (NT * KT);
  const int l = lane & 15, hi = lane >> 4;
  const int K = kt * 32 + hi * 8 + (j < 8 ? j : j + 8);
  const int N = nt * 16 + l;
  dst[idx] = (__bf16)src[((size_t)h * Kdim + K) * Ndim + N];
}

// V ([SEQ][DMOD] fp32, head-interleaved) -> per-head B-fragments (K=t, N=d)
__global__ void repack_v_kernel(const float* __restrict__ vsrc,
                                __bf16* __restrict__ dst) {
  const int idx = blockIdx.x * blockDim.x + threadIdx.x;  // NH*SEQ*DHEAD
  if (idx >= NH * SEQ * DHEAD) return;
  const int j = idx & 15;
  const int lane = (idx >> 4) & 31;
  const int f = idx >> 9;       // fragment id
  const int nt = f & 7;         // NT = DHEAD/16 = 8
  const int kt = (f >> 3) & 63; // KT = SEQ/32  = 64
  const int h = f >> 9;
  const int l = lane & 15, hi = lane >> 4;
  const int t = kt * 32 + hi * 8 + (j < 8 ? j : j + 8);
  const int d = nt * 16 + l;
  dst[idx] = (__bf16)vsrc[(size_t)t * DMOD + h * DHEAD + d];
}

// ---------------------------------------------------------------------------
// Feature-map kernel: per wave one (head, 16-row) block.
// Qf = |gelu(gelu(X Wq1) Wq2)|          -> A-layout frags
// Kf = |k2 + (k2 Wint)*sD2|, k2=|sD|*gelu(gelu(X Wk1) Wk2)  -> B-layout frags
// ---------------------------------------------------------------------------
__global__ __launch_bounds__(128) void feat_kernel(
    const float* __restrict__ q, const float* __restrict__ k,
    const float* __restrict__ sD, const float* __restrict__ sD2,
    const __bf16* __restrict__ w1q, const __bf16* __restrict__ w1k,
    const __bf16* __restrict__ w2q, const __bf16* __restrict__ w2k,
    const __bf16* __restrict__ wint, __bf16* __restrict__ qfrag,
    __bf16* __restrict__ kfrag) {
  __shared__ __bf16 lds[4][16 * 128];
  const int tid = threadIdx.x;
  const int wave = tid >> 5, lane = tid & 31;
  const int l = lane & 15, hi = lane >> 4;
  const int h = blockIdx.x >> 5;
  const int sb = (blockIdx.x & 31) * 4 + wave;  // 16-row block 0..127
  const int sbase = sb * 16;
  __bf16* T = lds[wave];

  const __bf16* w1qh = w1q + (size_t)h * DHEAD * DHID_;
  const __bf16* w1kh = w1k + (size_t)h * DHEAD * DHID_;
  const __bf16* w2qh = w2q + (size_t)h * DHID_ * DKER_;
  const __bf16* w2kh = w2k + (size_t)h * DHID_ * DKER_;
  const __bf16* winth = wint + (size_t)h * DKER_ * DKER_;

  v16bf xa[4], t1a[4];

  // ======================= Q path =======================
#pragma unroll
  for (int kt = 0; kt < 4; ++kt)
#pragma unroll
    for (int j = 0; j < 16; ++j) {
      const int K = kt * 32 + hi * 8 + (j < 8 ? j : j + 8);
      xa[kt][j] = (__bf16)q[(size_t)(sbase + l) * DMOD + h * DHEAD + K];
    }
  // T1 = gelu(X @ Wq1)  (16x128)
#pragma unroll
  for (int nt = 0; nt < 8; ++nt) {
    v8f acc = zero8();
#pragma unroll
    for (int kt = 0; kt < 4; ++kt) {
      v16bf b = *(const v16bf*)(w1qh + ((size_t)(kt * 8 + nt) * 32 + lane) * 16);
      acc = wmma_bf16(xa[kt], b, acc);
    }
#pragma unroll
    for (int j = 0; j < 8; ++j)
      T[(j + hi * 8) * 128 + nt * 16 + l] = (__bf16)gelu_exact(acc[j]);
  }
#pragma unroll
  for (int kt = 0; kt < 4; ++kt)
#pragma unroll
    for (int j = 0; j < 16; ++j)
      t1a[kt][j] = T[l * 128 + kt * 32 + hi * 8 + (j < 8 ? j : j + 8)];
  // T2 = |gelu(T1 @ Wq2)| (16x64) -> LDS
#pragma unroll
  for (int nt = 0; nt < 4; ++nt) {
    v8f acc = zero8();
#pragma unroll
    for (int kt = 0; kt < 4; ++kt) {
      v16bf b = *(const v16bf*)(w2qh + ((size_t)(kt * 4 + nt) * 32 + lane) * 16);
      acc = wmma_bf16(t1a[kt], b, acc);
    }
#pragma unroll
    for (int j = 0; j < 8; ++j)
      T[(j + hi * 8) * 64 + nt * 16 + l] = (__bf16)fabsf(gelu_exact(acc[j]));
  }
  // emit Qf fragments (A layout, lane = s row)
  __bf16* qfh = qfrag + (size_t)h * SEQ * DKER_;
#pragma unroll
  for (int kt = 0; kt < 2; ++kt) {
    v16bf o;
#pragma unroll
    for (int j = 0; j < 16; ++j)
      o[j] = T[l * 64 + kt * 32 + hi * 8 + (j < 8 ? j : j + 8)];
    *(v16bf*)(qfh + ((size_t)(sb * 2 + kt) * 32 + lane) * 16) = o;
  }

  // ======================= K path =======================
#pragma unroll
  for (int kt = 0; kt < 4; ++kt)
#pragma unroll
    for (int j = 0; j < 16; ++j) {
      const int K = kt * 32 + hi * 8 + (j < 8 ? j : j + 8);
      xa[kt][j] = (__bf16)k[(size_t)(sbase + l) * DMOD + h * DHEAD + K];
    }
#pragma unroll
  for (int nt = 0; nt < 8; ++nt) {
    v8f acc = zero8();
#pragma unroll
    for (int kt = 0; kt < 4; ++kt) {
      v16bf b = *(const v16bf*)(w1kh + ((size_t)(kt * 8 + nt) * 32 + lane) * 16);
      acc = wmma_bf16(xa[kt], b, acc);
    }
#pragma unroll
    for (int j = 0; j < 8; ++j)
      T[(j + hi * 8) * 128 + nt * 16 + l] = (__bf16)gelu_exact(acc[j]);
  }
#pragma unroll
  for (int kt = 0; kt < 4; ++kt)
#pragma unroll
    for (int j = 0; j < 16; ++j)
      t1a[kt][j] = T[l * 128 + kt * 32 + hi * 8 + (j < 8 ? j : j + 8)];
  // k2 = |sD| * gelu(T1 @ Wk2) (kept fp32 in C layout)
  v8f kc[4];
#pragma unroll
  for (int nt = 0; nt < 4; ++nt) {
    v8f acc = zero8();
#pragma unroll
    for (int kt = 0; kt < 4; ++kt) {
      v16bf b = *(const v16bf*)(w2kh + ((size_t)(kt * 4 + nt) * 32 + lane) * 16);
      acc = wmma_bf16(t1a[kt], b, acc);
    }
    const float scl = fabsf(sD[h * DKER_ + nt * 16 + l]);
#pragma unroll
    for (int j = 0; j < 8; ++j) kc[nt][j] = scl * gelu_exact(acc[j]);
  }
  // k2 -> LDS -> A frags for interaction GEMM
#pragma unroll
  for (int nt = 0; nt < 4; ++nt)
#pragma unroll
    for (int j = 0; j < 8; ++j)
      T[(j + hi * 8) * 64 + nt * 16 + l] = (__bf16)kc[nt][j];
  v16bf k2a[2];
#pragma unroll
  for (int kt = 0; kt < 2; ++kt)
#pragma unroll
    for (int j = 0; j < 16; ++j)
      k2a[kt][j] = T[l * 64 + kt * 32 + hi * 8 + (j < 8 ? j : j + 8)];
  // Kf = |k2 + (k2 @ Wint) * sD2| -> LDS
#pragma unroll
  for (int nt = 0; nt < 4; ++nt) {
    v8f acci = zero8();
#pragma unroll
    for (int kt = 0; kt < 2; ++kt) {
      v16bf b = *(const v16bf*)(winth + ((size_t)(kt * 4 + nt) * 32 + lane) * 16);
      acci = wmma_bf16(k2a[kt], b, acci);
    }
    const float s2 = sD2[h * DKER_ + nt * 16 + l];
#pragma unroll
    for (int j = 0; j < 8; ++j)
      T[(j + hi * 8) * 64 + nt * 16 + l] = (__bf16)fabsf(kc[nt][j] + acci[j] * s2);
  }
  // emit Kf fragments (B layout, lane = t column, K = hi*8 + (j<8 ? j : j+8))
  __bf16* kfh = kfrag + (size_t)h * SEQ * DKER_;
#pragma unroll
  for (int kt = 0; kt < 2; ++kt) {
    v16bf o;
#pragma unroll
    for (int j = 0; j < 16; ++j) o[j] = T[l * 64 + kt * 32 + hi * 8 + (j < 8 ? j : j + 8)];
    *(v16bf*)(kfh + ((size_t)(sb * 2 + kt) * 32 + lane) * 16) = o;
  }
}

// ---------------------------------------------------------------------------
// Attention kernel: one workgroup = (head, 16 s-rows); 4 waves split t-range.
// Single pass: U_s = sum_t [mask?(qk+eps):exp(sw)] v_t ; R_s = sum_t mask*qk
// out_s = U_s / (R_s + eps + exp(sparse_lse_s))
// ---------------------------------------------------------------------------
__global__ __launch_bounds__(128) void attn_kernel(
    const __bf16* __restrict__ qfrag, const __bf16* __restrict__ kfrag,
    const __bf16* __restrict__ vfrag, const unsigned char* __restrict__ mask,
    const float* __restrict__ sparse_lse, const float* __restrict__ sparse_w,
    float* __restrict__ out) {
  __shared__ __bf16 pbuf[4][16 * 32];
  __shared__ float redbuf[4][16 * DHEAD];
  __shared__ float rsumbuf[4][16];

  const int tid = threadIdx.x;
  const int wave = tid >> 5, lane = tid & 31;
  const int l = lane & 15, hi = lane >> 4;
  const int h = blockIdx.x >> 7;
  const int sb = blockIdx.x & 127;
  const int sbase = sb * 16;

  const __bf16* qf = qfrag + (size_t)h * SEQ * DKER_;
  const __bf16* kf = kfrag + (size_t)h * SEQ * DKER_;
  const __bf16* vfh = vfrag + (size_t)h * SEQ * DHEAD;
  const float* swh = sparse_w + (size_t)h * SEQ * SEQ;

  v16bf qa[2];
#pragma unroll
  for (int kt = 0; kt < 2; ++kt)
    qa[kt] = *(const v16bf*)(qf + ((size_t)(sb * 2 + kt) * 32 + lane) * 16);

  v8f acc[8];
#pragma unroll
  for (int nt = 0; nt < 8; ++nt) acc[nt] = zero8();
  float rsum[8];
#pragma unroll
  for (int j = 0; j < 8; ++j) rsum[j] = 0.0f;

  const int t0 = wave * (SEQ / 4);
  for (int tb32 = t0; tb32 < t0 + SEQ / 4; tb32 += 32) {
    if (tb32 + 32 < SEQ)  // prefetch next sparse-weights tile (global_prefetch_b8)
      __builtin_prefetch(swh + (size_t)(sbase + l) * SEQ + tb32 + 32, 0, 1);
#pragma unroll
    for (int tt = 0; tt < 2; ++tt) {
      const int tb = (tb32 >> 4) + tt;
      v8f sc = zero8();
#pragma unroll
      for (int kt = 0; kt < 2; ++kt) {
        v16bf bb = *(const v16bf*)(kf + ((size_t)(tb * 2 + kt) * 32 + lane) * 16);
        sc = wmma_bf16(qa[kt], bb, sc);
      }
      const int t = tb * 16 + l;
#pragma unroll
      for (int j = 0; j < 8; ++j) {
        const int srow = sbase + j + hi * 8;
        const size_t ix = (size_t)srow * SEQ + t;
        const bool m = mask[ix] != 0;
        const float sv = sc[j];
        const float comb = m ? (sv + EPSV) : expf(swh[ix]);
        rsum[j] += m ? sv : 0.0f;
        pbuf[wave][(j + hi * 8) * 32 + tt * 16 + l] = (__bf16)comb;
      }
    }
    // P tile -> A-layout fragment via LDS (same-wave, LDS is in-order)
    v16bf pa;
#pragma unroll
    for (int j = 0; j < 16; ++j)
      pa[j] = pbuf[wave][l * 32 + hi * 8 + (j < 8 ? j : j + 8)];
    const int ktv = tb32 >> 5;
#pragma unroll
    for (int nt = 0; nt < 8; ++nt) {
      v16bf vb = *(const v16bf*)(vfh + ((size_t)(ktv * 8 + nt) * 32 + lane) * 16);
      acc[nt] = wmma_bf16(pa, vb, acc[nt]);
    }
  }
  // reduce row-sums across the 16-lane halves (wave32 shfl)
#pragma unroll
  for (int j = 0; j < 8; ++j) {
    float r = rsum[j];
#pragma unroll
    for (int off = 1; off < 16; off <<= 1) r += __shfl_xor(r, off, 32);
    if (l == 0) rsumbuf[wave][hi * 8 + j] = r;
  }
#pragma unroll
  for (int nt = 0; nt < 8; ++nt)
#pragma unroll
    for (int j = 0; j < 8; ++j)
      redbuf[wave][(j + hi * 8) * DHEAD + nt * 16 + l] = acc[nt][j];
  __syncthreads();
  for (int i = tid; i < 16 * DHEAD; i += 128) {
    const int row = i >> 7, d = i & 127;
    const float u = redbuf[0][i] + redbuf[1][i] + redbuf[2][i] + redbuf[3][i];
    const float rs = rsumbuf[0][row] + rsumbuf[1][row] + rsumbuf[2][row] +
                     rsumbuf[3][row];
    const int s = sbase + row;
    const float Z = rs + EPSV + expf(sparse_lse[h * SEQ + s]);
    out[(size_t)s * DMOD + h * DHEAD + d] = u / Z;
  }
}

// ---------------------------------------------------------------------------
extern "C" void kernel_launch(void* const* d_in, const int* in_sizes, int n_in,
                              void* d_out, int out_size, void* d_ws,
                              size_t ws_size, hipStream_t stream) {
  (void)in_sizes; (void)n_in; (void)out_size; (void)ws_size;
  const float* q = (const float*)d_in[0];
  const float* k = (const float*)d_in[1];
  const float* v = (const float*)d_in[2];
  const unsigned char* mask = (const unsigned char*)d_in[3];  // jax bool = 1 byte
  const float* sparse_lse = (const float*)d_in[4];
  const float* sparse_w = (const float*)d_in[5];
  const float* kq1 = (const float*)d_in[6];
  const float* kk1 = (const float*)d_in[7];
  const float* kq2 = (const float*)d_in[8];
  const float* kk2 = (const float*)d_in[9];
  const float* wintm = (const float*)d_in[10];
  const float* sD = (const float*)d_in[11];
  const float* sD2 = (const float*)d_in[12];
  float* out = (float*)d_out;

  // workspace layout (~17.6 MiB total)
  char* ws = (char*)d_ws;
  size_t off = 0;
  auto take = [&](size_t bytes) -> char* {
    char* p = ws + off;
    off = (off + bytes + 255) & ~(size_t)255;
    return p;
  };
  __bf16* qfrag = (__bf16*)take((size_t)NH * SEQ * DKER_ * 2);
  __bf16* kfrag = (__bf16*)take((size_t)NH * SEQ * DKER_ * 2);
  __bf16* vfrag = (__bf16*)take((size_t)NH * SEQ * DHEAD * 2);
  __bf16* w1qf = (__bf16*)take((size_t)NH * DHEAD * DHID_ * 2);
  __bf16* w1kf = (__bf16*)take((size_t)NH * DHEAD * DHID_ * 2);
  __bf16* w2qf = (__bf16*)take((size_t)NH * DHID_ * DKER_ * 2);
  __bf16* w2kf = (__bf16*)take((size_t)NH * DHID_ * DKER_ * 2);
  __bf16* wintf = (__bf16*)take((size_t)NH * DKER_ * DKER_ * 2);

  auto cdiv = [](int a, int b) { return (a + b - 1) / b; };

  repack_b_kernel<<<cdiv(NH * DHEAD * DHID_, 256), 256, 0, stream>>>(
      kq1, w1qf, DHEAD, DHID_, NH);
  repack_b_kernel<<<cdiv(NH * DHEAD * DHID_, 256), 256, 0, stream>>>(
      kk1, w1kf, DHEAD, DHID_, NH);
  repack_b_kernel<<<cdiv(NH * DHID_ * DKER_, 256), 256, 0, stream>>>(
      kq2, w2qf, DHID_, DKER_, NH);
  repack_b_kernel<<<cdiv(NH * DHID_ * DKER_, 256), 256, 0, stream>>>(
      kk2, w2kf, DHID_, DKER_, NH);
  repack_b_kernel<<<cdiv(NH * DKER_ * DKER_, 256), 256, 0, stream>>>(
      wintm, wintf, DKER_, DKER_, NH);
  repack_v_kernel<<<cdiv(NH * SEQ * DHEAD, 256), 256, 0, stream>>>(v, vfrag);

  feat_kernel<<<NH * 32, 128, 0, stream>>>(q, k, sD, sD2, w1qf, w1kf, w2qf,
                                           w2kf, wintf, qfrag, kfrag);
  attn_kernel<<<NH * 128, 128, 0, stream>>>(qfrag, kfrag, vfrag, mask,
                                            sparse_lse, sparse_w, out);
}
